// CfCCell_51531017617706
// MI455X (gfx1250) — matrix operand order repaired and checked by reading, not compiled, not run on hardware
//
#include <hip/hip_runtime.h>
#include <math.h>

typedef __attribute__((ext_vector_type(16))) _Float16 v16h;
typedef __attribute__((ext_vector_type(8)))  _Float16 v8h;
typedef __attribute__((ext_vector_type(8)))  float    v8f;

#define HDIM 256
#define IDIM 128
#define BT   64            // rows per workgroup tile
#define LDX  (IDIM + 8)    // padded LDS row strides (halves)
#define LDH  (HDIM + 8)

// Branch-free erf, Abramowitz-Stegun 7.1.26 (|err| <= 1.5e-7), using
// v_rcp_f32 / v_exp_f32 hardware transcendentals. ~13 VALU ops.
static __device__ __forceinline__ float fast_erf(float x) {
    float ax = __builtin_fabsf(x);
    float t  = __builtin_amdgcn_rcpf(__builtin_fmaf(0.3275911f, ax, 1.0f));
    float p  = __builtin_fmaf(1.061405429f, t, -1.453152027f);
    p = __builtin_fmaf(p, t, 1.421413741f);
    p = __builtin_fmaf(p, t, -0.284496736f);
    p = __builtin_fmaf(p, t, 0.254829592f);
    p = p * t;
    float e = __builtin_amdgcn_exp2f(-ax * ax * 1.4426950408889634f);
    float r = __builtin_fmaf(-p, e, 1.0f);
    return __builtin_copysignf(r, x);
}

static __device__ __forceinline__ float gelu_exact(float v) {
    return 0.5f * v * (1.0f + fast_erf(v * 0.70710678118654752440f));
}

static __device__ __forceinline__ float fast_sigmoid(float z) {
    float e = __builtin_amdgcn_exp2f(-z * 1.4426950408889634f);
    return __builtin_amdgcn_rcpf(1.0f + e);
}

static __device__ __forceinline__ void zero_acc(v8f acc[4][2]) {
    v8f z = {0.f, 0.f, 0.f, 0.f, 0.f, 0.f, 0.f, 0.f};
#pragma unroll
    for (int mt = 0; mt < 4; ++mt) { acc[mt][0] = z; acc[mt][1] = z; }
}

// Accumulate acc += A(64xK from LDS, rows mt*16..) x W[nCol0.., kOff..]^T
// per-wave: 4 M-tiles x 2 N-tiles, K = KT*32.
// k order used here (not AMD's table order; valid since A and B match):
//  A 16x32 f16: lane = g*16 + m (g=lane>>4), halves = K in {g*16 .. g*16+15} (contiguous)
//  B 32x16 f16: lane = g*16 + n,             halves = K in {g*16 .. g*16+15} (contiguous)
template<int KT>
static __device__ __forceinline__ void gemm_tiles(
    v8f acc[4][2], const _Float16* shA, int ldA,
    const _Float16* __restrict__ W, int ldW, int kOff,
    int nCol0, int lane)
{
    const int g  = lane >> 4;
    const int nr = lane & 15;
#pragma unroll
    for (int kk = 0; kk < KT; ++kk) {
        const int kb = kOff + kk * 32 + g * 16;
        v16h b0 = *(const v16h*)(W + (size_t)(nCol0 +      nr) * ldW + kb);
        v16h b1 = *(const v16h*)(W + (size_t)(nCol0 + 16 + nr) * ldW + kb);
#pragma unroll
        for (int mt = 0; mt < 4; ++mt) {
            const _Float16* pa = shA + (mt * 16 + nr) * ldA + kk * 32 + g * 16;
            v8h lo = *(const v8h*)(pa);
            v8h hi = *(const v8h*)(pa + 8);
            v16h a;
#pragma unroll
            for (int i = 0; i < 8; ++i) { a[i] = lo[i]; a[8 + i] = hi[i]; }
            acc[mt][0] = __builtin_amdgcn_wmma_f32_16x16x32_f16(
                false, a, false, b0, (short)0, acc[mt][0], false, false);
            acc[mt][1] = __builtin_amdgcn_wmma_f32_16x16x32_f16(
                false, a, false, b1, (short)0, acc[mt][1], false, false);
        }
    }
}

__global__ void f32_to_f16_kernel(const float* __restrict__ src,
                                  _Float16* __restrict__ dst, int n) {
    int i = blockIdx.x * blockDim.x + threadIdx.x;
    if (i < n) dst[i] = (_Float16)src[i];
}

__global__ void __launch_bounds__(256)
cfc_fused_kernel(const float* __restrict__ x, const float* __restrict__ hx,
                 const _Float16* __restrict__ W1,  const _Float16* __restrict__ W2,
                 const _Float16* __restrict__ Wta, const _Float16* __restrict__ Wtb,
                 const _Float16* __restrict__ Win, const _Float16* __restrict__ Wr,
                 const float* __restrict__ b1,  const float* __restrict__ b2,
                 const float* __restrict__ bta, const float* __restrict__ btb,
                 const float* __restrict__ bin, const float* __restrict__ binp,
                 const float* __restrict__ br,  float* __restrict__ out, int Btot)
{
    __shared__ __align__(16) _Float16 shx [BT][LDX];
    __shared__ __align__(16) _Float16 shh [BT][LDH];
    __shared__ __align__(16) _Float16 shf1[BT][LDH];
    __shared__ __align__(16) _Float16 shf2[BT][LDH];

    const int tid   = threadIdx.x;
    const int lane  = tid & 31;
    const int wave  = tid >> 5;
    const int g     = lane >> 4;
    const int nr    = lane & 15;
    const int nCol0 = wave * 32;               // 8 waves x 32 cols = 256
    const size_t rowBase = (size_t)blockIdx.x * BT;

    // ---- stage x (64x128) and hx (64x256) into LDS as f16 ----
#pragma unroll
    for (int it = 0; it < (BT * IDIM / 4) / 256; ++it) {
        int idx = it * 256 + tid;              // idx = r*32 + c4
        int r = idx >> 5; int c = (idx & 31) * 4;
        float4 v = ((const float4*)x)[rowBase * (IDIM / 4) + idx];
        shx[r][c + 0] = (_Float16)v.x; shx[r][c + 1] = (_Float16)v.y;
        shx[r][c + 2] = (_Float16)v.z; shx[r][c + 3] = (_Float16)v.w;
    }
#pragma unroll
    for (int it = 0; it < (BT * HDIM / 4) / 256; ++it) {
        int idx = it * 256 + tid;              // idx = r*64 + c4
        int r = idx >> 6; int c = (idx & 63) * 4;
        float4 v = ((const float4*)hx)[rowBase * (HDIM / 4) + idx];
        shh[r][c + 0] = (_Float16)v.x; shh[r][c + 1] = (_Float16)v.y;
        shh[r][c + 2] = (_Float16)v.z; shh[r][c + 3] = (_Float16)v.w;
    }
    __syncthreads();

    v8f acc[4][2], accB[4][2], tv[4][2], sv[4][2];

    // ---- ff1 = gelu([x|hx] @ W1^T + b1) -> shf1 ----
    zero_acc(acc);
    gemm_tiles<4>(acc, &shx[0][0], LDX, W1, IDIM + HDIM, 0,    nCol0, lane);
    gemm_tiles<8>(acc, &shh[0][0], LDH, W1, IDIM + HDIM, IDIM, nCol0, lane);
#pragma unroll
    for (int j = 0; j < 2; ++j) {
        float bb = b1[nCol0 + j * 16 + nr];
#pragma unroll
        for (int mt = 0; mt < 4; ++mt)
#pragma unroll
            for (int r = 0; r < 8; ++r)
                shf1[mt * 16 + g * 8 + r][nCol0 + j * 16 + nr] =
                    (_Float16)gelu_exact(acc[mt][j][r] + bb);
    }
    __syncthreads();

    // ---- ff2 = gelu(ff1 @ W2^T + b2) -> shf2 ----
    zero_acc(acc);
    gemm_tiles<8>(acc, &shf1[0][0], LDH, W2, HDIM, 0, nCol0, lane);
#pragma unroll
    for (int j = 0; j < 2; ++j) {
        float bb = b2[nCol0 + j * 16 + nr];
#pragma unroll
        for (int mt = 0; mt < 4; ++mt)
#pragma unroll
            for (int r = 0; r < 8; ++r)
                shf2[mt * 16 + g * 8 + r][nCol0 + j * 16 + nr] =
                    (_Float16)gelu_exact(acc[mt][j][r] + bb);
    }
    __syncthreads();

    // ---- t = sigmoid((ff2@Wta^T + bta)*TS + ff2@Wtb^T + btb), TS = 1.0 ----
    zero_acc(acc);
    gemm_tiles<8>(acc,  &shf2[0][0], LDH, Wta, HDIM, 0, nCol0, lane);
    zero_acc(accB);
    gemm_tiles<8>(accB, &shf2[0][0], LDH, Wtb, HDIM, 0, nCol0, lane);
#pragma unroll
    for (int j = 0; j < 2; ++j) {
        float ba = bta[nCol0 + j * 16 + nr];
        float bb = btb[nCol0 + j * 16 + nr];
#pragma unroll
        for (int mt = 0; mt < 4; ++mt)
#pragma unroll
            for (int r = 0; r < 8; ++r) {
                float z = (acc[mt][j][r] + ba) * 1.0f + (accB[mt][j][r] + bb);
                tv[mt][j][r] = fast_sigmoid(z);
            }
    }

    // ---- s = gelu(x@Win^T + bin + input_b) + gelu(hx@Wr^T + r_b) ----
    zero_acc(acc);
    gemm_tiles<4>(acc, &shx[0][0], LDX, Win, IDIM, 0, nCol0, lane);
#pragma unroll
    for (int j = 0; j < 2; ++j) {
        float bb = bin[nCol0 + j * 16 + nr] + binp[nCol0 + j * 16 + nr];
#pragma unroll
        for (int mt = 0; mt < 4; ++mt)
#pragma unroll
            for (int r = 0; r < 8; ++r)
                sv[mt][j][r] = gelu_exact(acc[mt][j][r] + bb);
    }
    zero_acc(acc);
    gemm_tiles<8>(acc, &shh[0][0], LDH, Wr, HDIM, 0, nCol0, lane);
#pragma unroll
    for (int j = 0; j < 2; ++j) {
        float bb = br[nCol0 + j * 16 + nr];
#pragma unroll
        for (int mt = 0; mt < 4; ++mt)
#pragma unroll
            for (int r = 0; r < 8; ++r)
                sv[mt][j][r] += gelu_exact(acc[mt][j][r] + bb);
    }

    // ---- new_h = hx*(1-t) + t*s ; tuple output duplicated ----
    const size_t ofs2 = (size_t)Btot * HDIM;
#pragma unroll
    for (int j = 0; j < 2; ++j) {
        const int col = nCol0 + j * 16 + nr;
#pragma unroll
        for (int mt = 0; mt < 4; ++mt)
#pragma unroll
            for (int r = 0; r < 8; ++r) {
                size_t row = rowBase + (size_t)(mt * 16 + g * 8 + r);
                float hv = hx[row * HDIM + col];            // full-precision state
                float t  = tv[mt][j][r];
                float s  = sv[mt][j][r];
                float nh = hv * (1.0f - t) + t * s;
                out[row * HDIM + col]        = nh;
                out[ofs2 + row * HDIM + col] = nh;
            }
    }
}

extern "C" void kernel_launch(void* const* d_in, const int* in_sizes, int n_in,
                              void* d_out, int out_size, void* d_ws, size_t ws_size,
                              hipStream_t stream) {
    (void)n_in; (void)out_size; (void)ws_size;
    const float* x    = (const float*)d_in[0];
    const float* hx   = (const float*)d_in[1];
    const float* W1   = (const float*)d_in[2];
    const float* b1   = (const float*)d_in[3];
    const float* W2   = (const float*)d_in[4];
    const float* b2   = (const float*)d_in[5];
    const float* Wta  = (const float*)d_in[6];
    const float* bta  = (const float*)d_in[7];
    const float* Wtb  = (const float*)d_in[8];
    const float* btb  = (const float*)d_in[9];
    const float* Win  = (const float*)d_in[10];
    const float* bin  = (const float*)d_in[11];
    const float* binp = (const float*)d_in[12];
    const float* Wr   = (const float*)d_in[13];
    const float* br   = (const float*)d_in[14];
    float* out = (float*)d_out;

    const int Btot = in_sizes[0] / IDIM;       // 131072

    // f16 weight mirrors in workspace (768 KB total, L2-resident)
    _Float16* w1h  = (_Float16*)d_ws;
    _Float16* w2h  = w1h  + HDIM * (IDIM + HDIM);
    _Float16* wtah = w2h  + HDIM * HDIM;
    _Float16* wtbh = wtah + HDIM * HDIM;
    _Float16* winh = wtbh + HDIM * HDIM;
    _Float16* wrh  = winh + HDIM * IDIM;

    struct { const float* s; _Float16* d; int n; } cv[6] = {
        { W1,  w1h,  HDIM * (IDIM + HDIM) },
        { W2,  w2h,  HDIM * HDIM },
        { Wta, wtah, HDIM * HDIM },
        { Wtb, wtbh, HDIM * HDIM },
        { Win, winh, HDIM * IDIM },
        { Wr,  wrh,  HDIM * HDIM },
    };
    for (int i = 0; i < 6; ++i)
        f32_to_f16_kernel<<<(cv[i].n + 255) / 256, 256, 0, stream>>>(cv[i].s, cv[i].d, cv[i].n);

    cfc_fused_kernel<<<dim3(Btot / BT), dim3(256), 0, stream>>>(
        x, hx, w1h, w2h, wtah, wtbh, winh, wrh,
        b1, b2, bta, btb, bin, binp, br, out, Btot);
}
